// TEN_Encoder_33114197852603
// MI455X (gfx1250) — matrix operand order repaired and checked by reading, not compiled, not run on hardware
//
#include <hip/hip_runtime.h>
#include <hip/hip_bf16.h>

// ---------------- fixed problem dims (from reference) ----------------
constexpr int Bsz = 8, Tt = 2048, Dd = 512, Cc = 4, Kk = 16, Ll = 6;
constexpr int D2 = 2 * Dd;                 // 1024
constexpr int M_TOT = Bsz * Tt;            // 16384 rows

#define ENABLE_ASYNC 1                     // async global->LDS staging

// ---------------- WMMA types ----------------
typedef __attribute__((ext_vector_type(8)))  __bf16 v8bf;
typedef __attribute__((ext_vector_type(16))) __bf16 v16bf;
typedef __attribute__((ext_vector_type(8)))  float  v8f;

__device__ inline unsigned int f2bf(float f) {
  unsigned int u = __float_as_uint(f);
  unsigned int r = u + 0x7FFFu + ((u >> 16) & 1u);   // round-to-nearest-even
  return r >> 16;
}

__device__ inline void zero16(unsigned short* dst) {
  uint4 z; z.x = z.y = z.z = z.w = 0u;
  *(uint4*)dst = z; *(uint4*)(dst + 8) = z;
}

// 16B global -> LDS copy. Async path uses the CDNA5 async-DMA instruction:
// per-lane 16 bytes, LDS address = low 32 bits of the generic pointer
// (LDS aperture truncation, ISA 10.2), tracked by ASYNCcnt.
__device__ inline void cp16B(const unsigned short* __restrict__ g, unsigned short* l) {
#if ENABLE_ASYNC
  unsigned lds_off = (unsigned)(unsigned long long)(const void*)l;
  asm volatile("global_load_async_to_lds_b128 %0, %1, off"
               :: "v"(lds_off), "v"(g) : "memory");
#else
  *(uint4*)l = *(const uint4*)g;
#endif
}

__device__ inline void wait_async_lds() {
#if ENABLE_ASYNC
#if __has_builtin(__builtin_amdgcn_s_wait_asynccnt)
  __builtin_amdgcn_s_wait_asynccnt(0);
#else
  asm volatile("s_wait_asynccnt 0x0" ::: "memory");
#endif
#endif
}

// A fragment (16x32 bf16): lane (g=lane>>4, l=lane&15) holds row l,
// halves 0..15 -> K = 16g..16g+15 (matches B fragment order)
__device__ inline v16bf load_frag_a(const unsigned short* rowBase, int fg) {
  union { v16bf v; v8bf h[2]; } u;
  u.h[0] = *(const v8bf*)(rowBase + 16 * fg);
  u.h[1] = *(const v8bf*)(rowBase + 16 * fg + 8);
  return u.v;
}
// B fragment (32x16 bf16): lane holds column l, halves 0..15 -> K = 16g..16g+15
__device__ inline v16bf load_frag_b(const unsigned short* colBase, int fg) {
  union { v16bf v; v8bf h[2]; } u;
  const v8bf* p = (const v8bf*)(colBase + 16 * fg);
  u.h[0] = p[0]; u.h[1] = p[1];
  return u.v;
}

__device__ inline float wave_sum(float v) {
  #pragma unroll
  for (int off = 16; off > 0; off >>= 1) v += __shfl_xor(v, off, 32);
  return v;
}

// bulk f32 -> bf16 conversion (8 elems/thread); n must be a multiple of 2048
__global__ __launch_bounds__(256) void cvt_kernel(
    const float* __restrict__ src, unsigned short* __restrict__ dst, int n)
{
  const int i = (blockIdx.x * 256 + threadIdx.x) * 8;
  if (i >= n) return;
  const float4* p = (const float4*)(src + i);
  float4 a = p[0], b = p[1];
  uint4 s;
  s.x = f2bf(a.x) | (f2bf(a.y) << 16);
  s.y = f2bf(a.z) | (f2bf(a.w) << 16);
  s.z = f2bf(b.x) | (f2bf(b.y) << 16);
  s.w = f2bf(b.z) | (f2bf(b.w) << 16);
  *(uint4*)(dst + i) = s;
}

// =====================================================================
// WMMA GEMM on bf16: Out[M,N] = A[M,K] * Bw^T  (Bw row-major (N,K), bf16)
//   tile 128x128, 8 waves (4M x 2N), 32x64/wave, K-step 32,
//   async double-buffered LDS staging, rows padded to 40 halves (80B
//   = 20 banks -> conflict-free b128 fragment reads).
// EPI: 0 = +bias +pos_emb, writes f32 out + bf16 mirror outb
//      1 = beta scatter to [t,c,b,k]
//      2 = +bias, f32 out
// =====================================================================
constexpr int BM = 128, BN = 128, BK = 32, BKP = 40;

template <int EPI>
__global__ __launch_bounds__(256) void gemm_wmma(
    const unsigned short* __restrict__ A, const unsigned short* __restrict__ Bw,
    const float* __restrict__ bias, const float* __restrict__ aux,
    float* __restrict__ out, unsigned short* __restrict__ outb,
    int M, int N, int K)
{
  __shared__ __align__(16) unsigned short As[2][BM][BKP];   // 2 x 10 KB
  __shared__ __align__(16) unsigned short Bs[2][BN][BKP];   // 2 x 10 KB

  const int tid   = threadIdx.x;
  const int lane  = tid & 31, wid = tid >> 5;
  const int waveM = wid & 3,  waveN = wid >> 2;
  const int tileN = blockIdx.x * BN;
  const int tileM = blockIdx.y * BM;
  const int lr = tid >> 1;            // tile row 0..127 (staging)
  const int lh = (tid & 1) * 16;      // k-offset 0 / 16
  const int fg = lane >> 4, fl = lane & 15;

  const v8f vz = {0.f,0.f,0.f,0.f,0.f,0.f,0.f,0.f};
  v8f acc[2][4];
  #pragma unroll
  for (int i = 0; i < 2; ++i)
    #pragma unroll
    for (int j = 0; j < 4; ++j) acc[i][j] = vz;

  auto stage = [&](int buf, int k0) {
    const int gm = tileM + lr;                       // M multiple of 128
    cp16B(A + (size_t)gm * K + k0 + lh,     &As[buf][lr][lh]);
    cp16B(A + (size_t)gm * K + k0 + lh + 8, &As[buf][lr][lh + 8]);
    const int gn = tileN + lr;
    if (gn < N) {
      cp16B(Bw + (size_t)gn * K + k0 + lh,     &Bs[buf][lr][lh]);
      cp16B(Bw + (size_t)gn * K + k0 + lh + 8, &Bs[buf][lr][lh + 8]);
    } else {
      zero16(&Bs[buf][lr][lh]);
    }
  };

  stage(0, 0);
  wait_async_lds();
  __syncthreads();

  int buf = 0;
  for (int k0 = 0; k0 < K; k0 += BK) {
    if (k0 + BK < K) stage(buf ^ 1, k0 + BK);        // overlap with compute

    v16bf af[2], bfv[4];
    #pragma unroll
    for (int fm = 0; fm < 2; ++fm)
      af[fm] = load_frag_a(&As[buf][waveM * 32 + fm * 16 + fl][0], fg);
    #pragma unroll
    for (int fn = 0; fn < 4; ++fn)
      bfv[fn] = load_frag_b(&Bs[buf][waveN * 64 + fn * 16 + fl][0], fg);
    #pragma unroll
    for (int fm = 0; fm < 2; ++fm)
      #pragma unroll
      for (int fn = 0; fn < 4; ++fn)
        acc[fm][fn] = __builtin_amdgcn_wmma_f32_16x16x32_bf16(
            false, af[fm], false, bfv[fn], (short)0, acc[fm][fn], false, false);

    wait_async_lds();
    __syncthreads();
    buf ^= 1;
  }

  #pragma unroll
  for (int fm = 0; fm < 2; ++fm) {
    const int mBase = tileM + waveM * 32 + fm * 16 + 8 * fg;
    #pragma unroll
    for (int fn = 0; fn < 4; ++fn) {
      const int n = tileN + waveN * 64 + fn * 16 + fl;
      if (n >= N) continue;
      v8f a = acc[fm][fn];
      #pragma unroll
      for (int i = 0; i < 8; ++i) {
        const int m = mBase + i;
        const float val = a[i];
        if constexpr (EPI == 0) {
          const float o = val + bias[n] + aux[(size_t)(m & (Tt - 1)) * N + n];
          out[(size_t)m * N + n] = o;
          outb[(size_t)m * N + n] = (unsigned short)f2bf(o);
        } else if constexpr (EPI == 1) {
          const int bb = m >> 11, t = m & (Tt - 1);
          const int cc = n >> 4,  kq = n & 15;
          out[(size_t)((t * Cc + cc) * Bsz + bb) * Kk + kq] = val;
        } else {
          out[(size_t)m * N + n] = val + bias[n];
        }
      }
    }
  }
}

// =====================================================================
// Fused gate/up GEMM (bf16 A/B): epilogue silu(g+bg)*(u+bu) -> bf16
// tile 128x64, 8 waves (4M x 2N), 32x32/wave per matrix, async staged.
// =====================================================================
__global__ __launch_bounds__(256) void gemm_gateup(
    const unsigned short* __restrict__ A, const unsigned short* __restrict__ Wgw,
    const unsigned short* __restrict__ Wuw, const float* __restrict__ bg,
    const float* __restrict__ bu, unsigned short* __restrict__ guOut,
    int M, int N, int K)
{
  __shared__ __align__(16) unsigned short As[2][128][BKP];
  __shared__ __align__(16) unsigned short Gs[2][64][BKP];
  __shared__ __align__(16) unsigned short Us[2][64][BKP];

  const int tid = threadIdx.x;
  const int lane = tid & 31, wid = tid >> 5;
  const int waveM = wid & 3, waveN = wid >> 2;
  const int tileN = blockIdx.x * 64;
  const int tileM = blockIdx.y * 128;
  const int fg = lane >> 4, fl = lane & 15;

  const v8f vz = {0.f,0.f,0.f,0.f,0.f,0.f,0.f,0.f};
  v8f accg[2][2], accu[2][2];
  #pragma unroll
  for (int i = 0; i < 2; ++i)
    #pragma unroll
    for (int j = 0; j < 2; ++j) { accg[i][j] = vz; accu[i][j] = vz; }

  auto stage = [&](int buf, int k0) {
    const int lr = tid >> 1, lh = (tid & 1) * 16;
    cp16B(A + (size_t)(tileM + lr) * K + k0 + lh,     &As[buf][lr][lh]);
    cp16B(A + (size_t)(tileM + lr) * K + k0 + lh + 8, &As[buf][lr][lh + 8]);
    const int t2 = tid & 127;
    const int nr = t2 >> 1, lh2 = (t2 & 1) * 16;
    const unsigned short* W = (tid < 128) ? Wgw : Wuw;
    unsigned short* dst = (tid < 128) ? &Gs[buf][nr][lh2] : &Us[buf][nr][lh2];
    cp16B(W + (size_t)(tileN + nr) * K + k0 + lh2,     dst);
    cp16B(W + (size_t)(tileN + nr) * K + k0 + lh2 + 8, dst + 8);
  };

  stage(0, 0);
  wait_async_lds();
  __syncthreads();

  int buf = 0;
  for (int k0 = 0; k0 < K; k0 += BK) {
    if (k0 + BK < K) stage(buf ^ 1, k0 + BK);

    v16bf af[2], gf[2], uf[2];
    #pragma unroll
    for (int fm = 0; fm < 2; ++fm)
      af[fm] = load_frag_a(&As[buf][waveM * 32 + fm * 16 + fl][0], fg);
    #pragma unroll
    for (int fn = 0; fn < 2; ++fn) {
      gf[fn] = load_frag_b(&Gs[buf][waveN * 32 + fn * 16 + fl][0], fg);
      uf[fn] = load_frag_b(&Us[buf][waveN * 32 + fn * 16 + fl][0], fg);
    }
    #pragma unroll
    for (int fm = 0; fm < 2; ++fm)
      #pragma unroll
      for (int fn = 0; fn < 2; ++fn) {
        accg[fm][fn] = __builtin_amdgcn_wmma_f32_16x16x32_bf16(
            false, af[fm], false, gf[fn], (short)0, accg[fm][fn], false, false);
        accu[fm][fn] = __builtin_amdgcn_wmma_f32_16x16x32_bf16(
            false, af[fm], false, uf[fn], (short)0, accu[fm][fn], false, false);
      }

    wait_async_lds();
    __syncthreads();
    buf ^= 1;
  }

  #pragma unroll
  for (int fm = 0; fm < 2; ++fm) {
    const int mBase = tileM + waveM * 32 + fm * 16 + 8 * fg;
    #pragma unroll
    for (int fn = 0; fn < 2; ++fn) {
      const int n = tileN + waveN * 32 + fn * 16 + fl;
      v8f g = accg[fm][fn], u = accu[fm][fn];
      #pragma unroll
      for (int i = 0; i < 8; ++i) {
        const int m = mBase + i;
        float gv = g[i] + bg[n];
        float uv = u[i] + bu[n];
        float s  = gv / (1.0f + __expf(-gv));      // silu
        guOut[(size_t)m * D2 + n] = (unsigned short)f2bf(s * uv);
      }
    }
  }
}

// =====================================================================
// Temporal scan: 16 blocks x wave32; 2 (c,b) groups per wave, 16-wide
// shuffle matvec with R columns in registers.  beta/rs: [t][c][b][k].
// =====================================================================
__global__ __launch_bounds__(32) void scan_kernel(
    const float* __restrict__ beta, const float* __restrict__ alpha,
    const float* __restrict__ omega, const float* __restrict__ Rm,
    float* __restrict__ rs)
{
  const int tid = threadIdx.x;
  const int k = tid & 15;
  const int p = blockIdx.x * 2 + (tid >> 4);     // 0..31 == c*8 + b
  const int c = p >> 3;

  const float mag = 1.0f / (1.0f + __expf(-alpha[c * Kk + k]));
  const float ang = omega[c * Kk + k];           // DT == 1
  const float cph = __cosf(ang), sph = __sinf(ang);
  float Rcol[Kk];
  #pragma unroll
  for (int kk = 0; kk < Kk; ++kk) Rcol[kk] = Rm[(c * Kk + kk) * Kk + k];

  float r = 0.f, im = 0.f;
  for (int t = 0; t < Tt; ++t) {
    const float bt = beta[(size_t)t * (Cc * Bsz * Kk) + p * Kk + k];
    const float nr = mag * (r * cph - im * sph) + bt;
    const float ni = mag * (r * sph + im * cph);
    float rr = 0.f, ii = 0.f;
    #pragma unroll
    for (int kk = 0; kk < Kk; ++kk) {
      rr += __shfl(nr, kk, 16) * Rcol[kk];
      ii += __shfl(ni, kk, 16) * Rcol[kk];
    }
    r = rr; im = ii;
    rs[(size_t)t * (Cc * Bsz * Kk) + p * Kk + k] = r;
  }
}

// =====================================================================
// cell_out = rs @ Wout^T + bout, per-cell LN over D, mean over cells.
// wave per (b,t) row; lane covers 16 of 512 columns.
// =====================================================================
__global__ __launch_bounds__(256) void cellmix_kernel(
    const float* __restrict__ rs, const float* __restrict__ Wout,
    const float* __restrict__ bout, const float* __restrict__ cg,
    const float* __restrict__ cb, float* __restrict__ mixed)
{
  const int lane = threadIdx.x & 31;
  const int row  = blockIdx.x * 8 + (threadIdx.x >> 5);   // m = b*T + t
  const int b = row >> 11, t = row & (Tt - 1);

  float acc[16];
  #pragma unroll
  for (int j = 0; j < 16; ++j) acc[j] = 0.f;

  for (int c = 0; c < Cc; ++c) {
    float rsv[Kk];
    const float* rp = rs + (size_t)((t * Cc + c) * Bsz + b) * Kk;
    #pragma unroll
    for (int kk = 0; kk < Kk; ++kk) rsv[kk] = rp[kk];     // broadcast load

    float yv[16], s = 0.f, s2 = 0.f;
    #pragma unroll
    for (int j = 0; j < 16; ++j) {
      const int d = lane + 32 * j;
      const float* wp = Wout + (size_t)(c * Dd + d) * Kk;
      float y = bout[c * Dd + d];
      #pragma unroll
      for (int kk = 0; kk < Kk; ++kk) y += rsv[kk] * wp[kk];
      yv[j] = y; s += y; s2 += y * y;
    }
    s = wave_sum(s); s2 = wave_sum(s2);
    const float mean = s * (1.0f / Dd);
    const float var  = s2 * (1.0f / Dd) - mean * mean;
    const float inv  = rsqrtf(var + 1e-5f);
    #pragma unroll
    for (int j = 0; j < 16; ++j) {
      const int d = lane + 32 * j;
      acc[j] += ((yv[j] - mean) * inv * cg[c * Dd + d] + cb[c * Dd + d]) * 0.25f;
    }
  }
  #pragma unroll
  for (int j = 0; j < 16; ++j)
    mixed[(size_t)row * Dd + lane + 32 * j] = acc[j];
}

// =====================================================================
// h = LN(h + add) (add may be null); also writes bf16 mirror hb.
// =====================================================================
__global__ __launch_bounds__(256) void ln_add_kernel(
    float* __restrict__ h, unsigned short* __restrict__ hb,
    const float* __restrict__ add,
    const float* __restrict__ g, const float* __restrict__ bb)
{
  const int lane = threadIdx.x & 31;
  const int row  = blockIdx.x * 8 + (threadIdx.x >> 5);

  float v[16], s = 0.f, s2 = 0.f;
  #pragma unroll
  for (int j = 0; j < 16; ++j) {
    const int d = lane + 32 * j;
    float x = h[(size_t)row * Dd + d];
    if (add) x += add[(size_t)row * Dd + d];
    v[j] = x; s += x; s2 += x * x;
  }
  s = wave_sum(s); s2 = wave_sum(s2);
  const float mean = s * (1.0f / Dd);
  const float var  = s2 * (1.0f / Dd) - mean * mean;
  const float inv  = rsqrtf(var + 1e-5f);
  #pragma unroll
  for (int j = 0; j < 16; ++j) {
    const int d = lane + 32 * j;
    const float o = (v[j] - mean) * inv * g[d] + bb[d];
    h[(size_t)row * Dd + d] = o;
    hb[(size_t)row * Dd + d] = (unsigned short)f2bf(o);
  }
}

// mean pooling over T
__global__ __launch_bounds__(256) void pool_kernel(
    const float* __restrict__ h, float* __restrict__ pooled)
{
  const int b = blockIdx.x;
  for (int rep = 0; rep < 2; ++rep) {
    const int d = threadIdx.x + rep * 256;
    float s = 0.f;
    for (int t = 0; t < Tt; ++t)
      s += h[((size_t)b * Tt + t) * Dd + d];
    pooled[b * Dd + d] = s * (1.0f / Tt);
  }
}

// classifier head: gelu(pooled@W1^T+b1)@W2^T+b2  (block per batch row)
__global__ __launch_bounds__(256) void head_kernel(
    const float* __restrict__ pooled, const float* __restrict__ W1,
    const float* __restrict__ b1, const float* __restrict__ W2,
    const float* __restrict__ b2, float* __restrict__ out)
{
  __shared__ float hid[256];
  const int b = blockIdx.x, j = threadIdx.x;
  float s = b1[j];
  const float* pr = pooled + (size_t)b * Dd;
  const float* wr = W1 + (size_t)j * Dd;
  for (int d = 0; d < Dd; ++d) s += pr[d] * wr[d];
  hid[j] = 0.5f * s * (1.0f + erff(s * 0.70710678118654752f));
  __syncthreads();
  if (j < 2) {
    float o = b2[j];
    const float* w2 = W2 + (size_t)j * 256;
    for (int q = 0; q < 256; ++q) o += hid[q] * w2[q];
    out[b * 2 + j] = o;
  }
}

// =====================================================================
extern "C" void kernel_launch(void* const* d_in, const int* in_sizes, int n_in,
                              void* d_out, int out_size, void* d_ws, size_t ws_size,
                              hipStream_t stream)
{
  const float* x    = (const float*)d_in[0];
  const float* Win  = (const float*)d_in[1];
  const float* bin_ = (const float*)d_in[2];
  const float* pos  = (const float*)d_in[3];
  const float* alpha= (const float*)d_in[4];
  const float* omega= (const float*)d_in[5];
  const float* cWin = (const float*)d_in[6];
  const float* Rm   = (const float*)d_in[7];
  const float* Wout = (const float*)d_in[8];
  const float* bout = (const float*)d_in[9];
  const float* cg   = (const float*)d_in[10];
  const float* cb   = (const float*)d_in[11];
  const float* ln1g = (const float*)d_in[12];
  const float* ln1b = (const float*)d_in[13];
  const float* Wg   = (const float*)d_in[14];
  const float* bg   = (const float*)d_in[15];
  const float* Wu   = (const float*)d_in[16];
  const float* bu   = (const float*)d_in[17];
  const float* Wd   = (const float*)d_in[18];
  const float* bd   = (const float*)d_in[19];
  const float* ln2g = (const float*)d_in[20];
  const float* ln2b = (const float*)d_in[21];
  const float* ng   = (const float*)d_in[22];
  const float* nb   = (const float*)d_in[23];
  const float* W1   = (const float*)d_in[24];
  const float* b1   = (const float*)d_in[25];
  const float* W2   = (const float*)d_in[26];
  const float* b2   = (const float*)d_in[27];
  float* out = (float*)d_out;

  // ---- workspace carve-out (~140 MB) ----
  char* ws = (char*)d_ws;
  size_t off = 0;
  float* h     = (float*)(ws + off); off += (size_t)M_TOT * Dd * 4;              // 32 MB
  float* mixed = (float*)(ws + off); off += (size_t)M_TOT * Dd * 4;              // 32 MB
  unsigned short* gu = (unsigned short*)(ws + off); off += (size_t)M_TOT * D2 * 2; // 32 MB
  unsigned short* hb = (unsigned short*)(ws + off); off += (size_t)M_TOT * Dd * 2; // 16 MB
  float* beta  = (float*)(ws + off); off += (size_t)Tt * Cc * Bsz * Kk * 4;      // 4 MB
  float* rs    = (float*)(ws + off); off += (size_t)Tt * Cc * Bsz * Kk * 4;      // 4 MB
  float* pooled= (float*)(ws + off); off += (size_t)Bsz * Dd * 4;
  off = (off + 255) & ~(size_t)255;
  unsigned short* Win_b  = (unsigned short*)(ws + off); off += (size_t)Dd * Dd * 2;
  unsigned short* cWin_b = (unsigned short*)(ws + off); off += (size_t)Ll * Cc * Kk * Dd * 2;
  unsigned short* Wg_b   = (unsigned short*)(ws + off); off += (size_t)Ll * D2 * Dd * 2;
  unsigned short* Wu_b   = (unsigned short*)(ws + off); off += (size_t)Ll * D2 * Dd * 2;
  unsigned short* Wd_b   = (unsigned short*)(ws + off); off += (size_t)Ll * Dd * D2 * 2;
  unsigned short* xb     = gu;     // gu unused until first gate/up; reuse for x(bf16)

  const dim3 blk(256);
  const dim3 gRows(M_TOT / 8);  // wave-per-row kernels

  // ---- one-time bf16 conversions (weights + input) ----
  auto cvt = [&](const float* s, unsigned short* d, int n) {
    cvt_kernel<<<dim3(n / 2048), blk, 0, stream>>>(s, d, n);
  };
  cvt(x,    xb,     M_TOT * Dd);
  cvt(Win,  Win_b,  Dd * Dd);
  cvt(cWin, cWin_b, Ll * Cc * Kk * Dd);
  cvt(Wg,   Wg_b,   Ll * D2 * Dd);
  cvt(Wu,   Wu_b,   Ll * D2 * Dd);
  cvt(Wd,   Wd_b,   Ll * Dd * D2);

  // h = x @ Win^T + bin + pos   (also writes hb bf16 mirror)
  gemm_wmma<0><<<dim3(Dd / BN, M_TOT / BM), blk, 0, stream>>>(
      xb, Win_b, bin_, pos, h, hb, M_TOT, Dd, Dd);

  for (int l = 0; l < Ll; ++l) {
    // beta[t,c,b,k] = h @ cWin[l]^T   (N = C*K = 64, padded to 128-tile)
    gemm_wmma<1><<<dim3(1, M_TOT / BM), blk, 0, stream>>>(
        hb, cWin_b + (size_t)l * Cc * Kk * Dd, nullptr, nullptr,
        beta, nullptr, M_TOT, Cc * Kk, Dd);

    // recurrent scan over T
    scan_kernel<<<dim3(16), dim3(32), 0, stream>>>(
        beta, alpha + l * Cc * Kk, omega + l * Cc * Kk,
        Rm + (size_t)l * Cc * Kk * Kk, rs);

    // per-cell out-proj + LN + mean over cells
    cellmix_kernel<<<gRows, blk, 0, stream>>>(
        rs, Wout + (size_t)l * Cc * Dd * Kk, bout + (size_t)l * Cc * Dd,
        cg + (size_t)l * Cc * Dd, cb + (size_t)l * Cc * Dd, mixed);

    // h = LN(h + mixed)  (+ hb mirror)
    ln_add_kernel<<<gRows, blk, 0, stream>>>(h, hb, mixed, ln1g + l * Dd, ln1b + l * Dd);

    // gu = silu(h@Wg^T+bg) * (h@Wu^T+bu)  (bf16)
    gemm_gateup<<<dim3(D2 / 64, M_TOT / 128), blk, 0, stream>>>(
        hb, Wg_b + (size_t)l * D2 * Dd, Wu_b + (size_t)l * D2 * Dd,
        bg + (size_t)l * D2, bu + (size_t)l * D2, gu, M_TOT, D2, Dd);

    // mixed = gu @ Wd^T + bd
    gemm_wmma<2><<<dim3(Dd / BN, M_TOT / BM), blk, 0, stream>>>(
        gu, Wd_b + (size_t)l * Dd * D2, bd + (size_t)l * Dd, nullptr,
        mixed, nullptr, M_TOT, Dd, D2);

    // h = LN(h + mixed)  (+ hb mirror)
    ln_add_kernel<<<gRows, blk, 0, stream>>>(h, hb, mixed, ln2g + l * Dd, ln2b + l * Dd);
  }

  // final norm, pool, head
  ln_add_kernel<<<gRows, blk, 0, stream>>>(h, hb, nullptr, ng, nb);
  pool_kernel<<<dim3(Bsz), blk, 0, stream>>>(h, pooled);
  head_kernel<<<dim3(Bsz), blk, 0, stream>>>(pooled, W1, b1, W2, b2, out);
}
